// MambaBackbone_1717986918812
// MI455X (gfx1250) — compile-verified, matrix operand order checked by reading, not run on hardware
//
#include <hip/hip_runtime.h>

// ---------------------------------------------------------------------------
// Mamba backbone for MI455X (gfx1250, wave32, WMMA, async LDS staging)
// ---------------------------------------------------------------------------
#define BQ      4
#define LQ      1024
#define DIN     8
#define DMODEL  512
#define DEPTH   4
#define DINNER  1024
#define DSTATE  16
#define DCONV   4
#define DTRANK  32
#define BL      (BQ * LQ)        // 4096 tokens

typedef __bf16 bf16;
typedef __attribute__((ext_vector_type(16))) __bf16 bf16x16;
typedef __attribute__((ext_vector_type(8)))  __bf16 bf16x8;
typedef __attribute__((ext_vector_type(8)))  float  f32x8;
typedef __attribute__((ext_vector_type(4)))  int    i32x4;

#if __has_builtin(__builtin_amdgcn_global_load_async_to_lds_b128)
#define USE_ASYNC_LDS 1
#else
#define USE_ASYNC_LDS 0
#endif

__device__ __forceinline__ void wait_async_zero()
{
#if USE_ASYNC_LDS
#if __has_builtin(__builtin_amdgcn_s_wait_asynccnt)
    __builtin_amdgcn_s_wait_asynccnt(0);
#else
    asm volatile("s_wait_asynccnt 0x0" ::: "memory");
#endif
#endif
}

// ---------------------------------------------------------------------------
// Generic bf16 WMMA GEMM:  C[M,N] = A[M,K] @ W[N,K]^T   (f32 accumulate)
//
// Block = 8 waves; all waves share n0, so the 64x32 B tile (4 KB) is staged
// ONCE per block into LDS via async global->LDS b128 (ASYNCcnt), double
// buffered so the next stage overlaps the current WMMAs. Each wave computes
// TWO 16-row M tiles against the staged B tile: 8 WMMAs per k-step for
// 8 ds_load_b128 (B reused 2x, A fragments reused 4x). A fragments stay on
// the direct global->VGPR path (+prefetch).
//
// A-fragment layout (16-bit A 16x32): lanes 0-15 rows m, K = 0..7 & 16..23;
// lanes 16-31 rows m, K = 8..15 & 24..31. B mirrors with lane = column.
// LDS B tile rows padded to 40 elements (80 B) for conflict-free b128 reads.
// Requires M%256==0, N%64==0, K%32==0 (true for every GEMM in this net).
// ---------------------------------------------------------------------------
__global__ __launch_bounds__(256) void wmma_gemm_bf16(
    const bf16* __restrict__ A, int lda,
    const bf16* __restrict__ W, int ldb,
    float* __restrict__ C, int ldc,
    int M, int N, int K)
{
    constexpr int BSTRIDE = 40;                    // padded row stride (elems)
    __shared__ bf16 Btile[2][64 * BSTRIDE];        // 2 x 5 KB

    const int lane = threadIdx.x & 31;
    const int wave = threadIdx.x >> 5;
    const int hl   = lane >> 4;        // K-half select (0 or 1)
    const int r16  = lane & 15;        // row (A) / column (B) within tile

    const int m0 = (blockIdx.y * 8 + wave) * 32;   // two 16-row tiles per wave
    const int n0 = blockIdx.x * 64;
    if (m0 >= M) return;               // never taken for our launches

    // B staging map: thread t moves 16 B; row = t>>2 (0..63), chunk = t&3
    const int   brow   = threadIdx.x >> 2;
    const int   bchunk = (threadIdx.x & 3) * 8;    // element offset 0,8,16,24
    const bf16* bSrc   = W + (size_t)(n0 + brow) * ldb + bchunk;
    const int   bOff   = brow * BSTRIDE + bchunk;

#if USE_ASYNC_LDS
    typedef __attribute__((address_space(1))) i32x4 gi32x4_t;
    typedef __attribute__((address_space(3))) i32x4 li32x4_t;
#define STAGE_B(k0, buf)                                                     \
    __builtin_amdgcn_global_load_async_to_lds_b128(                          \
        (gi32x4_t*)(bSrc + (k0)), (li32x4_t*)&Btile[(buf)][bOff], 0, 0)
#else
#define STAGE_B(k0, buf)                                                     \
    *(bf16x8*)&Btile[(buf)][bOff] = *(const bf16x8*)(bSrc + (k0))
#endif

    STAGE_B(0, 0);                                 // prologue stage

    f32x8 accA0 = {}, accA1 = {}, accA2 = {}, accA3 = {};   // M tile 0
    f32x8 accB0 = {}, accB1 = {}, accB2 = {}, accB3 = {};   // M tile 1
    const bf16* aRow0 = A + (size_t)(m0      + r16) * lda + hl * 8;
    const bf16* aRow1 = A + (size_t)(m0 + 16 + r16) * lda + hl * 8;

    for (int k0 = 0; k0 < K; k0 += 32) {
        const int buf = (k0 >> 5) & 1;

        // A fragments: direct global -> VGPR (unique per wave)
        union FragA { bf16x16 v; bf16x8 h[2]; } a0, a1;
        a0.h[0] = *(const bf16x8*)(aRow0 + k0);
        a0.h[1] = *(const bf16x8*)(aRow0 + k0 + 16);
        a1.h[0] = *(const bf16x8*)(aRow1 + k0);
        a1.h[1] = *(const bf16x8*)(aRow1 + k0 + 16);
        if (k0 + 32 < K) {
            __builtin_prefetch(aRow0 + k0 + 32, 0, 0);  // global_prefetch_b8
            __builtin_prefetch(aRow1 + k0 + 32, 0, 0);
        }

        wait_async_zero();             // Btile[buf] writes landed (this wave)
        __syncthreads();               // ... and for every wave in the block

        if (k0 + 32 < K) STAGE_B(k0 + 32, buf ^ 1);     // overlapped stage

        // B fragments from LDS (ds_load_b128 x2 each, conflict-free),
        // each reused by both M tiles (2 WMMAs per fragment).
        union FragB { bf16x16 v; bf16x8 h[2]; } b;
        {
            const bf16* p = &Btile[buf][(0 + r16) * BSTRIDE + hl * 8];
            b.h[0] = *(const bf16x8*)p;  b.h[1] = *(const bf16x8*)(p + 16);
            accA0 = __builtin_amdgcn_wmma_f32_16x16x32_bf16(
                        false, a0.v, false, b.v, (short)0, accA0, false, false);
            accB0 = __builtin_amdgcn_wmma_f32_16x16x32_bf16(
                        false, a1.v, false, b.v, (short)0, accB0, false, false);
        }
        {
            const bf16* p = &Btile[buf][(16 + r16) * BSTRIDE + hl * 8];
            b.h[0] = *(const bf16x8*)p;  b.h[1] = *(const bf16x8*)(p + 16);
            accA1 = __builtin_amdgcn_wmma_f32_16x16x32_bf16(
                        false, a0.v, false, b.v, (short)0, accA1, false, false);
            accB1 = __builtin_amdgcn_wmma_f32_16x16x32_bf16(
                        false, a1.v, false, b.v, (short)0, accB1, false, false);
        }
        {
            const bf16* p = &Btile[buf][(32 + r16) * BSTRIDE + hl * 8];
            b.h[0] = *(const bf16x8*)p;  b.h[1] = *(const bf16x8*)(p + 16);
            accA2 = __builtin_amdgcn_wmma_f32_16x16x32_bf16(
                        false, a0.v, false, b.v, (short)0, accA2, false, false);
            accB2 = __builtin_amdgcn_wmma_f32_16x16x32_bf16(
                        false, a1.v, false, b.v, (short)0, accB2, false, false);
        }
        {
            const bf16* p = &Btile[buf][(48 + r16) * BSTRIDE + hl * 8];
            b.h[0] = *(const bf16x8*)p;  b.h[1] = *(const bf16x8*)(p + 16);
            accA3 = __builtin_amdgcn_wmma_f32_16x16x32_bf16(
                        false, a0.v, false, b.v, (short)0, accA3, false, false);
            accB3 = __builtin_amdgcn_wmma_f32_16x16x32_bf16(
                        false, a1.v, false, b.v, (short)0, accB3, false, false);
        }

        __syncthreads();               // reads of Btile[buf] retired
    }
#undef STAGE_B

    // C layout: VGPR r -> row base + hl*8 + r, column n0 + r16 (+16j per tile)
    float* c0 = C + (size_t)(m0      + hl * 8) * ldc + n0 + r16;
    float* c1 = C + (size_t)(m0 + 16 + hl * 8) * ldc + n0 + r16;
#pragma unroll
    for (int r = 0; r < 8; ++r) {
        c0[(size_t)r * ldc +  0] = accA0[r];
        c0[(size_t)r * ldc + 16] = accA1[r];
        c0[(size_t)r * ldc + 32] = accA2[r];
        c0[(size_t)r * ldc + 48] = accA3[r];
        c1[(size_t)r * ldc +  0] = accB0[r];
        c1[(size_t)r * ldc + 16] = accB1[r];
        c1[(size_t)r * ldc + 32] = accB2[r];
        c1[(size_t)r * ldc + 48] = accB3[r];
    }
}

// ---------------------------------------------------------------------------
// Input projection: h[bl,dm] = x[bl,:8] . in_w[dm,:8] + in_b[dm]
// ---------------------------------------------------------------------------
__global__ __launch_bounds__(256) void in_proj_kernel(
    const float* __restrict__ x, const float* __restrict__ w,
    const float* __restrict__ bias,
    float* __restrict__ h, bf16* __restrict__ hb)
{
    int idx = blockIdx.x * 256 + threadIdx.x;
    if (idx >= BL * DMODEL) return;
    int bl = idx >> 9;
    int dm = idx & (DMODEL - 1);
    const float* xr = x + (size_t)bl * DIN;
    const float* wr = w + (size_t)dm * DIN;
    float acc = bias[dm];
#pragma unroll
    for (int i = 0; i < DIN; ++i) acc += xr[i] * wr[i];
    h[idx]  = acc;
    hb[idx] = (bf16)acc;
}

// ---------------------------------------------------------------------------
// Causal depthwise conv (k=4) + bias + SiLU.  xc lives in xz[:, 0:DINNER].
// ---------------------------------------------------------------------------
__global__ __launch_bounds__(256) void conv_silu_kernel(
    const float* __restrict__ xz, const float* __restrict__ cw,
    const float* __restrict__ cb,
    float* __restrict__ u, bf16* __restrict__ ub)
{
    int idx = blockIdx.x * 256 + threadIdx.x;
    if (idx >= BL * DINNER) return;
    int bl = idx >> 10;
    int d  = idx & (DINNER - 1);
    int l  = bl & (LQ - 1);
    const float* cwd = cw + (size_t)d * DCONV;
    float acc = cb[d];
#pragma unroll
    for (int k = 0; k < DCONV; ++k) {
        int lp = l + k - (DCONV - 1);
        if (lp >= 0)
            acc += xz[(size_t)(bl + k - (DCONV - 1)) * (2 * DINNER) + d] * cwd[k];
    }
    float s = acc / (1.0f + __expf(-acc));
    u[idx]  = s;
    ub[idx] = (bf16)s;
}

// ---------------------------------------------------------------------------
// dt = softplus(dt_lin + b_dt[d])   (in place on dt_lin)
// ---------------------------------------------------------------------------
__global__ __launch_bounds__(256) void softplus_bias_kernel(
    float* __restrict__ dt, const float* __restrict__ bdt)
{
    int idx = blockIdx.x * 256 + threadIdx.x;
    if (idx >= BL * DINNER) return;
    int d = idx & (DINNER - 1);
    float v = dt[idx] + bdt[d];
    dt[idx] = (v > 20.0f) ? v : __logf(1.0f + __expf(v));
}

// ---------------------------------------------------------------------------
// f32 -> bf16 elementwise convert
// ---------------------------------------------------------------------------
__global__ __launch_bounds__(256) void f32_to_bf16_kernel(
    const float* __restrict__ s, bf16* __restrict__ d, int n)
{
    int idx = blockIdx.x * 256 + threadIdx.x;
    if (idx < n) d[idx] = (bf16)s[idx];
}

// ---------------------------------------------------------------------------
// Selective scan. One thread per (b,d) channel: 16-wide state in VGPRs,
// per-timestep B/C (32 floats) staged in LDS (shared by the block's fixed b).
// Output fused with +u*D, *silu(z) gate, written as bf16 for the out-proj.
// ---------------------------------------------------------------------------
__global__ __launch_bounds__(256) void scan_kernel(
    const float* __restrict__ dt, const float* __restrict__ u,
    const float* __restrict__ xdbl, const float* __restrict__ xz,
    const float* __restrict__ A_log, const float* __restrict__ Dp,
    bf16* __restrict__ yb)
{
    int gid = blockIdx.x * 256 + threadIdx.x;     // B*DINNER threads total
    int b = gid >> 10;                            // DINNER == 1024
    int d = gid & (DINNER - 1);

    float A[DSTATE];
#pragma unroll
    for (int s = 0; s < DSTATE; ++s)
        A[s] = -__expf(A_log[(size_t)d * DSTATE + s]);
    const float Dv = Dp[d];

    float st[DSTATE];
#pragma unroll
    for (int s = 0; s < DSTATE; ++s) st[s] = 0.0f;

    __shared__ float bc[2 * DSTATE];              // B then C for current step

    for (int l = 0; l < LQ; ++l) {
        size_t bl = (size_t)b * LQ + l;
        if (threadIdx.x < 2 * DSTATE)
            bc[threadIdx.x] = xdbl[bl * (DTRANK + 2 * DSTATE) + DTRANK + threadIdx.x];
        __syncthreads();

        float dtv = dt[bl * DINNER + d];
        float uv  = u[bl * DINNER + d];
        float y = 0.0f;
#pragma unroll
        for (int s = 0; s < DSTATE; ++s) {
            float dA = __expf(dtv * A[s]);
            st[s] = dA * st[s] + dtv * bc[s] * uv;
            y += st[s] * bc[DSTATE + s];
        }
        float zv  = xz[bl * (2 * DINNER) + DINNER + d];
        float out = (y + uv * Dv) * (zv / (1.0f + __expf(-zv)));
        yb[bl * DINNER + d] = (bf16)out;
        __syncthreads();
    }
}

// ---------------------------------------------------------------------------
// LayerNorm over last dim (512).  One block per row, 2 elems per thread.
// ---------------------------------------------------------------------------
__global__ __launch_bounds__(256) void layernorm_kernel(
    const float* __restrict__ h, const float* __restrict__ g,
    const float* __restrict__ beta, float* __restrict__ out)
{
    int row = blockIdx.x;
    const float* hr = h + (size_t)row * DMODEL;
    __shared__ float red[256];

    float v0 = hr[threadIdx.x];
    float v1 = hr[threadIdx.x + 256];
    red[threadIdx.x] = v0 + v1;
    __syncthreads();
    for (int s = 128; s > 0; s >>= 1) {
        if (threadIdx.x < s) red[threadIdx.x] += red[threadIdx.x + s];
        __syncthreads();
    }
    float mu = red[0] * (1.0f / DMODEL);
    __syncthreads();

    float d0 = v0 - mu, d1 = v1 - mu;
    red[threadIdx.x] = d0 * d0 + d1 * d1;
    __syncthreads();
    for (int s = 128; s > 0; s >>= 1) {
        if (threadIdx.x < s) red[threadIdx.x] += red[threadIdx.x + s];
        __syncthreads();
    }
    float rs = rsqrtf(red[0] * (1.0f / DMODEL) + 1e-5f);

    float* orow = out + (size_t)row * DMODEL;
    orow[threadIdx.x]       = d0 * rs * g[threadIdx.x]       + beta[threadIdx.x];
    orow[threadIdx.x + 256] = d1 * rs * g[threadIdx.x + 256] + beta[threadIdx.x + 256];
}

// ---------------------------------------------------------------------------
// Host-side orchestration
// ---------------------------------------------------------------------------
extern "C" void kernel_launch(void* const* d_in, const int* in_sizes, int n_in,
                              void* d_out, int out_size, void* d_ws, size_t ws_size,
                              hipStream_t stream)
{
    const float* x      = (const float*)d_in[0];
    const float* in_w   = (const float*)d_in[1];
    const float* in_b   = (const float*)d_in[2];
    const float* W_xz   = (const float*)d_in[3];   // (4, 2048, 512)
    const float* conv_w = (const float*)d_in[4];   // (4, 1024, 4)
    const float* conv_b = (const float*)d_in[5];   // (4, 1024)
    const float* W_xp   = (const float*)d_in[6];   // (4, 64, 1024)
    const float* W_dt   = (const float*)d_in[7];   // (4, 1024, 32)
    const float* b_dt   = (const float*)d_in[8];   // (4, 1024)
    const float* A_log  = (const float*)d_in[9];   // (4, 1024, 16)
    const float* D_par  = (const float*)d_in[10];  // (4, 1024)
    const float* W_out  = (const float*)d_in[11];  // (4, 512, 1024)
    const float* ln_g   = (const float*)d_in[12];
    const float* ln_b   = (const float*)d_in[13];
    float* out = (float*)d_out;

    char* ws = (char*)d_ws;
    size_t off = 0;
    auto alloc = [&](size_t bytes) -> void* {
        void* p = ws + off;
        off += (bytes + 255) & ~(size_t)255;
        return p;
    };

    float* h     = (float*)alloc((size_t)BL * DMODEL * 4);
    bf16*  hb    = (bf16*) alloc((size_t)BL * DMODEL * 2);
    float* xz    = (float*)alloc((size_t)BL * 2 * DINNER * 4);
    float* u     = (float*)alloc((size_t)BL * DINNER * 4);
    bf16*  ub    = (bf16*) alloc((size_t)BL * DINNER * 2);
    float* xdbl  = (float*)alloc((size_t)BL * (DTRANK + 2 * DSTATE) * 4);
    bf16*  xdblb = (bf16*) alloc((size_t)BL * (DTRANK + 2 * DSTATE) * 2);
    float* dt    = (float*)alloc((size_t)BL * DINNER * 4);
    bf16*  yb    = (bf16*) alloc((size_t)BL * DINNER * 2);
    bf16*  wxzb  = (bf16*) alloc((size_t)DEPTH * 2 * DINNER * DMODEL * 2);
    bf16*  wxpb  = (bf16*) alloc((size_t)DEPTH * (DTRANK + 2 * DSTATE) * DINNER * 2);
    bf16*  wdtb  = (bf16*) alloc((size_t)DEPTH * DINNER * DTRANK * 2);
    bf16*  woutb = (bf16*) alloc((size_t)DEPTH * DMODEL * DINNER * 2);

    auto cvt = [&](const float* s, bf16* d, int n) {
        f32_to_bf16_kernel<<<(n + 255) / 256, 256, 0, stream>>>(s, d, n);
    };
    auto gemm = [&](const bf16* A, int lda, const bf16* W, int ldb,
                    float* C, int ldc, int M, int N, int K) {
        dim3 grid(N / 64, M / 256);        // wave covers 32 M rows now
        wmma_gemm_bf16<<<grid, 256, 0, stream>>>(A, lda, W, ldb, C, ldc, M, N, K);
    };

    // Weight conversions (all layers up front)
    cvt(W_xz,  wxzb,  DEPTH * 2 * DINNER * DMODEL);
    cvt(W_xp,  wxpb,  DEPTH * (DTRANK + 2 * DSTATE) * DINNER);
    cvt(W_dt,  wdtb,  DEPTH * DINNER * DTRANK);
    cvt(W_out, woutb, DEPTH * DMODEL * DINNER);

    // Input projection
    in_proj_kernel<<<(BL * DMODEL + 255) / 256, 256, 0, stream>>>(x, in_w, in_b, h, hb);

    for (int layer = 0; layer < DEPTH; ++layer) {
        const bf16*  wxz  = wxzb  + (size_t)layer * 2 * DINNER * DMODEL;
        const bf16*  wxp  = wxpb  + (size_t)layer * (DTRANK + 2 * DSTATE) * DINNER;
        const bf16*  wdt  = wdtb  + (size_t)layer * DINNER * DTRANK;
        const bf16*  wout = woutb + (size_t)layer * DMODEL * DINNER;
        const float* cw   = conv_w + (size_t)layer * DINNER * DCONV;
        const float* cb   = conv_b + (size_t)layer * DINNER;
        const float* bdt  = b_dt   + (size_t)layer * DINNER;
        const float* alog = A_log  + (size_t)layer * DINNER * DSTATE;
        const float* dp   = D_par  + (size_t)layer * DINNER;

        // xz = h @ W_xz^T          [4096 x 2048], K=512
        gemm(hb, DMODEL, wxz, DMODEL, xz, 2 * DINNER, BL, 2 * DINNER, DMODEL);

        // u = silu(conv(xc) + cb)
        conv_silu_kernel<<<(BL * DINNER + 255) / 256, 256, 0, stream>>>(xz, cw, cb, u, ub);

        // xdbl = u @ W_xp^T        [4096 x 64], K=1024
        gemm(ub, DINNER, wxp, DINNER, xdbl, DTRANK + 2 * DSTATE,
             BL, DTRANK + 2 * DSTATE, DINNER);
        cvt(xdbl, xdblb, BL * (DTRANK + 2 * DSTATE));

        // dt_lin = dt_lr @ W_dt^T  [4096 x 1024], K=32 (lda = 64: xdbl row)
        gemm(xdblb, DTRANK + 2 * DSTATE, wdt, DTRANK, dt, DINNER, BL, DINNER, DTRANK);
        softplus_bias_kernel<<<(BL * DINNER + 255) / 256, 256, 0, stream>>>(dt, bdt);

        // selective scan (fused + u*D and silu(z) gate) -> y (bf16)
        scan_kernel<<<(BQ * DINNER) / 256, 256, 0, stream>>>(
            dt, u, xdbl, xz, alog, dp, yb);

        // h = y @ W_out^T          [4096 x 512], K=1024
        gemm(yb, DINNER, wout, DINNER, h, DMODEL, BL, DMODEL, DINNER);
        cvt(h, hb, BL * DMODEL);
    }

    // Final LayerNorm
    layernorm_kernel<<<BL, 256, 0, stream>>>(h, ln_g, ln_b, out);
}
